// TransConvLayer_22419729285548
// MI455X (gfx1250) — matrix operand order repaired and checked by reading, not compiled, not run on hardware
//
#include <hip/hip_runtime.h>
#include <hip/hip_bf16.h>

// ---------------------------------------------------------------------------
// MI455X (gfx1250) implementation.
//   D=512, H=8, DK=64, N=L=4096.  All heavy math via v_wmma_f32_16x16x32_bf16.
// Pipeline:
//   0) transpose+convert weights to bf16  Wt[Nc][K]
//   1) value = src@Wv1+b   -> bf16 [L,512]
//   2) q     = qry@Wq+b    -> bf16 [N,512]
//   3) k     = src@Wk+b    -> bf16 [L,512]
//   4) v     = value@Wv2+b -> bf16 TRANSPOSED vt[H*512+d][L]  (B-operand ready)
//   5) fused attention: 32-row n-tile x 2048-row l-chunk per workgroup
//      (grid 128x2 = 256 WGs); wave h computes head-h scores (WMMA),
//      softmax over HEADS via LDS, each wave contracts all heads against its
//      64-wide d-slice (WMMA); l-partials combined with f32 global atomics.
// ---------------------------------------------------------------------------

typedef __attribute__((ext_vector_type(16))) __bf16 v16bf;
typedef __attribute__((ext_vector_type(8)))  __bf16 v8bf;
typedef __attribute__((ext_vector_type(8)))  float  v8f;

union AB { v16bf v; v8bf h[2]; };

#define DMODEL 512
#define HEADS  8
#define DKH    64
#define NTOT   4096
#define LTOT   4096
#define NTILE  32
#define LSPLIT 2
#define LCHUNK (LTOT / LSPLIT)

// ---------------------------------------------------------------------------
// Weight transpose + f32->bf16:  Wt[n][k] = (bf16) W[k][n]
// ---------------------------------------------------------------------------
__global__ void wtrans_kernel(const float* __restrict__ W, __bf16* __restrict__ Wt,
                              int K, int Nc) {
  size_t i = (size_t)blockIdx.x * blockDim.x + threadIdx.x;
  size_t tot = (size_t)K * Nc;
  if (i >= tot) return;
  int n = (int)(i / K), k = (int)(i % K);
  Wt[i] = (__bf16)W[(size_t)k * Nc + n];
}

__global__ void zero_kernel(float* __restrict__ p, int n) {
  int i = blockIdx.x * blockDim.x + threadIdx.x;
  if (i < n) p[i] = 0.f;
}

// ---------------------------------------------------------------------------
// GEMM: C[M,Nc] = A[M,K] @ Wt^T + bias   (Wt is [Nc][K] row-major == W^T)
// One wave per 64x64 output tile (16 f32 accumulators = 128 VGPRs); K-step 32.
// Each B tile is reused across 4 A tiles: 16 WMMAs per 8 x 32B loads per lane.
// ---------------------------------------------------------------------------
template <bool ABF16, bool TRANSOUT>
__global__ __launch_bounds__(32)
void gemm_bias_bf16_kernel(const void* __restrict__ Aab,
                           const __bf16* __restrict__ Wt,
                           const float* __restrict__ bias,
                           __bf16* __restrict__ Cout,
                           int M, int K, int Nc, int ldo) {
  const int lane  = threadIdx.x & 31;
  const int mbase = blockIdx.x * 64;
  const int nb    = blockIdx.y * 64;
  const int rc  = lane & 15;      // A-row / B-col / C-col sub-index
  const int hi  = lane >> 4;
  const int kb8 = hi * 8;         // A-operand K sub-block
  const int k16 = hi * 16;        // unused; B also uses kb8

  v8f acc[4][4];
#pragma unroll
  for (int mt = 0; mt < 4; ++mt)
#pragma unroll
    for (int t = 0; t < 4; ++t) acc[mt][t] = v8f{};

  for (int k0 = 0; k0 < K; k0 += 32) {
    AB am[4];
#pragma unroll
    for (int mt = 0; mt < 4; ++mt) {
      const int row = mbase + mt * 16 + rc;
      if (ABF16) {
        const __bf16* Ap = (const __bf16*)Aab + (size_t)row * K + k0;
        am[mt].h[0] = *(const v8bf*)(Ap + kb8);
        am[mt].h[1] = *(const v8bf*)(Ap + 16 + kb8);
      } else {
        const float* Af = (const float*)Aab + (size_t)row * K + k0;
        v8bf t0, t1;
#pragma unroll
        for (int j = 0; j < 8; ++j) {
          t0[j] = (__bf16)Af[kb8 + j];
          t1[j] = (__bf16)Af[16 + kb8 + j];
        }
        am[mt].h[0] = t0; am[mt].h[1] = t1;
      }
    }
#pragma unroll
    for (int t = 0; t < 4; ++t) {
      const __bf16* Bp = Wt + (size_t)(nb + t * 16 + rc) * K + k0 + kb8;
      AB b;
      b.h[0] = *(const v8bf*)(Bp);
      b.h[1] = *(const v8bf*)(Bp + 16);
#pragma unroll
      for (int mt = 0; mt < 4; ++mt)
        acc[mt][t] = __builtin_amdgcn_wmma_f32_16x16x32_bf16(
            false, am[mt].v, false, b.v, (short)0, acc[mt][t], false, false);
    }
  }

#pragma unroll
  for (int t = 0; t < 4; ++t) {
    const int ncol = nb + t * 16 + rc;
    const float bs = bias[ncol];
#pragma unroll
    for (int mt = 0; mt < 4; ++mt)
#pragma unroll
      for (int r = 0; r < 8; ++r) {
        const int Mi = mbase + mt * 16 + r + 8 * hi;
        const __bf16 o = (__bf16)(acc[mt][t][r] + bs);
        if (TRANSOUT) Cout[(size_t)ncol * ldo + Mi] = o;
        else          Cout[(size_t)Mi * ldo + ncol] = o;
      }
  }
}

// ---------------------------------------------------------------------------
// Fused GAT attention. Block = 256 threads = 8 waves (one wave per head for
// the score phase; one wave per 64-wide d-slice for the AV phase).
// 32 query rows per WG, 2-way l-split (softmax is over HEADS -> l-partials of
// the output are exact); partials combined via global_atomic_add_f32.
// ---------------------------------------------------------------------------
__global__ __launch_bounds__(256)
void gat_attn_kernel(const __bf16* __restrict__ qb,   // [N][512]
                     const __bf16* __restrict__ kbuf, // [L][512]
                     const __bf16* __restrict__ vt,   // [H*512][L] (transposed)
                     float* __restrict__ out) {       // [N][512], pre-zeroed
  __shared__ __align__(16) float  sS[HEADS][NTILE][32];  // raw scores  (32 KB)
  __shared__ __align__(16) __bf16 sW[HEADS][NTILE][32];  // softmax wts (16 KB)

  const int tid  = threadIdx.x;
  const int lane = tid & 31;
  const int wave = tid >> 5;           // == head in score phase, d-slice in AV
  const int nbase = blockIdx.x * NTILE;
  const int l0    = blockIdx.y * LCHUNK;
  const int rc  = lane & 15;
  const int hi  = lane >> 4;
  const int kb8 = hi * 8;
  const int k16 = hi * 16;

  // Q A-operands for head==wave: 2 n-halves x 2 K-chunks, kept in registers.
  AB qa[2][2];
#pragma unroll
  for (int nt = 0; nt < 2; ++nt) {
    const __bf16* Qp = qb + (size_t)(nbase + nt * 16 + rc) * DMODEL + wave * DKH;
#pragma unroll
    for (int c = 0; c < 2; ++c) {
      qa[nt][c].h[0] = *(const v8bf*)(Qp + c * 32 + kb8);
      qa[nt][c].h[1] = *(const v8bf*)(Qp + c * 32 + 16 + kb8);
    }
  }

  v8f oacc[2][4];
#pragma unroll
  for (int nt = 0; nt < 2; ++nt)
#pragma unroll
    for (int t = 0; t < 4; ++t) oacc[nt][t] = v8f{};

  for (int lbase = l0; lbase < l0 + LCHUNK; lbase += 32) {
    // Prefetch next l-tile of K and (this wave's V slice); wrapped address so
    // the loop stays branch-free (last-iteration prefetch is harmless).
    {
      const int lnext = l0 + ((lbase + 32 - l0) & (LCHUNK - 1));
      __builtin_prefetch(kbuf + (size_t)(lnext + rc) * DMODEL + wave * DKH, 0, 1);
      __builtin_prefetch(vt + (size_t)(wave * DKH + rc) * LTOT + lnext, 0, 1);
    }

    // ---- scores for head `wave`: S[32n x 32l] = 4 C-tiles, K=64 ----------
    v8f s[2][2] = {{v8f{}, v8f{}}, {v8f{}, v8f{}}};
#pragma unroll
    for (int half = 0; half < 2; ++half) {
      const __bf16* Kp =
          kbuf + (size_t)(lbase + half * 16 + rc) * DMODEL + wave * DKH + kb8;
      AB b[2];
#pragma unroll
      for (int c = 0; c < 2; ++c) {
        b[c].h[0] = *(const v8bf*)(Kp + c * 32);
        b[c].h[1] = *(const v8bf*)(Kp + c * 32 + 16);
      }
#pragma unroll
      for (int nt = 0; nt < 2; ++nt)
#pragma unroll
        for (int c = 0; c < 2; ++c)
          s[nt][half] = __builtin_amdgcn_wmma_f32_16x16x32_bf16(
              false, qa[nt][c].v, false, b[c].v, (short)0, s[nt][half],
              false, false);
    }
    const float inv_scale = 0.125f;  // 1/sqrt(DK)
#pragma unroll
    for (int nt = 0; nt < 2; ++nt)
#pragma unroll
      for (int r = 0; r < 8; ++r) {
        const int Mi = nt * 16 + r + 8 * hi;
        sS[wave][Mi][rc]      = s[nt][0][r] * inv_scale;
        sS[wave][Mi][16 + rc] = s[nt][1][r] * inv_scale;
      }
    __syncthreads();

    // ---- softmax over heads: 1024 (n,l) positions, 4 per thread -----------
#pragma unroll
    for (int u = 0; u < 4; ++u) {
      const int e  = tid * 4 + u;
      const int nn = e >> 5, ll = e & 31;
      float vv[HEADS], m = -3.0e38f;
#pragma unroll
      for (int h = 0; h < HEADS; ++h) { vv[h] = sS[h][nn][ll]; m = fmaxf(m, vv[h]); }
      float ssum = 0.f;
#pragma unroll
      for (int h = 0; h < HEADS; ++h) { vv[h] = __expf(vv[h] - m); ssum += vv[h]; }
      const float rs = 1.f / ssum;
#pragma unroll
      for (int h = 0; h < HEADS; ++h) sW[h][nn][ll] = (__bf16)(vv[h] * rs);
    }
    __syncthreads();

    // ---- AV: this wave owns out cols [wave*64, wave*64+64) over all heads -
#pragma unroll
    for (int h = 0; h < HEADS; ++h) {
      AB aw[2];  // attn weights as 16x32 bf16 A-operands, from LDS
#pragma unroll
      for (int nt = 0; nt < 2; ++nt) {
        aw[nt].h[0] = *(const v8bf*)&sW[h][nt * 16 + rc][kb8];
        aw[nt].h[1] = *(const v8bf*)&sW[h][nt * 16 + rc][16 + kb8];
      }
#pragma unroll
      for (int t = 0; t < 4; ++t) {
        const __bf16* Vp =
            vt + (size_t)(h * DMODEL + wave * DKH + t * 16 + rc) * LTOT + lbase + kb8;
        AB b;
        b.h[0] = *(const v8bf*)(Vp);
        b.h[1] = *(const v8bf*)(Vp + 16);
#pragma unroll
        for (int nt = 0; nt < 2; ++nt)
          oacc[nt][t] = __builtin_amdgcn_wmma_f32_16x16x32_bf16(
              false, aw[nt].v, false, b.v, (short)0, oacc[nt][t], false, false);
      }
    }
    // next-iteration barrier #1 protects sW against early overwrite
  }

  // mean over heads = *1/8; combine l-chunk partials with f32 atomics.
#pragma unroll
  for (int nt = 0; nt < 2; ++nt)
#pragma unroll
    for (int t = 0; t < 4; ++t)
#pragma unroll
      for (int r = 0; r < 8; ++r) {
        const int Mi = nbase + nt * 16 + r + 8 * hi;
        unsafeAtomicAdd(&out[(size_t)Mi * DMODEL + wave * DKH + t * 16 + rc],
                        oacc[nt][t][r] * 0.125f);
      }
}

// ---------------------------------------------------------------------------
extern "C" void kernel_launch(void* const* d_in, const int* in_sizes, int n_in,
                              void* d_out, int out_size, void* d_ws, size_t ws_size,
                              hipStream_t stream) {
  const float* query  = (const float*)d_in[0];
  const float* source = (const float*)d_in[1];
  const float* Wv1_w  = (const float*)d_in[2];
  const float* Wv1_b  = (const float*)d_in[3];
  const float* Wq_w   = (const float*)d_in[4];
  const float* Wq_b   = (const float*)d_in[5];
  const float* Wk_w   = (const float*)d_in[6];
  const float* Wk_b   = (const float*)d_in[7];
  const float* Wv2_w  = (const float*)d_in[8];
  const float* Wv2_b  = (const float*)d_in[9];
  float* out = (float*)d_out;

  // Workspace carve-up (bf16 everywhere): ~49.5 MB total
  char* ws = (char*)d_ws;
  __bf16* Wv1t = (__bf16*)ws;  ws += (size_t)DMODEL * DMODEL * 2;         // 0.5 MB
  __bf16* Wqt  = (__bf16*)ws;  ws += (size_t)DMODEL * DMODEL * 2;         // 0.5 MB
  __bf16* Wkt  = (__bf16*)ws;  ws += (size_t)DMODEL * DMODEL * 2;         // 0.5 MB
  __bf16* Wv2t = (__bf16*)ws;  ws += (size_t)DMODEL * (HEADS*DMODEL) * 2; // 4 MB
  __bf16* valb = (__bf16*)ws;  ws += (size_t)LTOT * DMODEL * 2;           // 4 MB
  __bf16* qbuf = (__bf16*)ws;  ws += (size_t)NTOT * DMODEL * 2;           // 4 MB
  __bf16* kbuf = (__bf16*)ws;  ws += (size_t)LTOT * DMODEL * 2;           // 4 MB
  __bf16* vtb  = (__bf16*)ws;  ws += (size_t)(HEADS*DMODEL) * LTOT * 2;   // 32 MB

  // 0) weights -> bf16, transposed to [Nc][K]; zero the atomic output buffer
  {
    int totA = DMODEL * DMODEL;
    wtrans_kernel<<<totA / 256, 256, 0, stream>>>(Wv1_w, Wv1t, DMODEL, DMODEL);
    wtrans_kernel<<<totA / 256, 256, 0, stream>>>(Wq_w,  Wqt,  DMODEL, DMODEL);
    wtrans_kernel<<<totA / 256, 256, 0, stream>>>(Wk_w,  Wkt,  DMODEL, DMODEL);
    int totB = DMODEL * HEADS * DMODEL;
    wtrans_kernel<<<totB / 256, 256, 0, stream>>>(Wv2_w, Wv2t, DMODEL, HEADS * DMODEL);
    int totO = NTOT * DMODEL;
    zero_kernel<<<totO / 256, 256, 0, stream>>>(out, totO);
  }

  // 1-3) projections (f32 activations in, bf16 out); 64x64 tile per wave
  gemm_bias_bf16_kernel<false, false>
      <<<dim3(LTOT / 64, DMODEL / 64), 32, 0, stream>>>(
          source, Wv1t, Wv1_b, valb, LTOT, DMODEL, DMODEL, DMODEL);
  gemm_bias_bf16_kernel<false, false>
      <<<dim3(NTOT / 64, DMODEL / 64), 32, 0, stream>>>(
          query, Wqt, Wq_b, qbuf, NTOT, DMODEL, DMODEL, DMODEL);
  gemm_bias_bf16_kernel<false, false>
      <<<dim3(LTOT / 64, DMODEL / 64), 32, 0, stream>>>(
          source, Wkt, Wk_b, kbuf, LTOT, DMODEL, DMODEL, DMODEL);

  // 4) v = value @ Wv2 + b, stored TRANSPOSED: vt[h*512+d][l]
  gemm_bias_bf16_kernel<true, true>
      <<<dim3(LTOT / 64, (HEADS * DMODEL) / 64), 32, 0, stream>>>(
          valb, Wv2t, Wv2_b, vtb, LTOT, DMODEL, HEADS * DMODEL, LTOT);

  // 5) fused attention + head-mean (128 n-tiles x 2 l-chunks = 256 WGs)
  gat_attn_kernel<<<dim3(NTOT / NTILE, LSPLIT), 256, 0, stream>>>(
      qbuf, kbuf, vtb, out);
}
